// GatedLinearAttention_32839319945671
// MI455X (gfx1250) — compile-verified, matrix operand order checked by reading, not run on hardware
//
#include <hip/hip_runtime.h>
#include <hip/hip_bf16.h>

typedef __bf16 bf16;
typedef __attribute__((ext_vector_type(16))) __bf16 v16bf;
typedef __attribute__((ext_vector_type(8)))  __bf16 v8bf;
typedef __attribute__((ext_vector_type(8)))  float  v8f;

#define HID 1024
#define NH 16
#define HD 64
#define RANK 16
#define BB 2
#define LL 2048
#define NTOK (BB * LL)        // 4096
#define N4   (4 * HID)        // fused Q|K|V|Gpre output width = 4096

// ---------------------------------------------------------------------------
// f32 -> bf16 elementwise convert (activations)
// ---------------------------------------------------------------------------
__global__ void cvt_f32_bf16_kernel(const float* __restrict__ in,
                                    bf16* __restrict__ out, int n) {
    int i = blockIdx.x * blockDim.x + threadIdx.x;
    if (i < n) out[i] = (bf16)in[i];
}

// ---------------------------------------------------------------------------
// Tiled transpose + convert: W (R x C, f32, row-major) -> WT (C x R, bf16).
// 32x32 tiles via LDS (33-col pad kills bank conflicts). Block (32,8).
// ---------------------------------------------------------------------------
__global__ void __launch_bounds__(256)
transpose_cvt_kernel(const float* __restrict__ W, bf16* __restrict__ WT,
                     int R, int C) {
    __shared__ float tile[32][33];
    const int c0 = blockIdx.x * 32;
    const int r0 = blockIdx.y * 32;
    const int tx = threadIdx.x;        // 0..31
    const int ty = threadIdx.y;        // 0..7
#pragma unroll
    for (int j = 0; j < 32; j += 8)
        tile[ty + j][tx] = W[(size_t)(r0 + ty + j) * C + c0 + tx];
    __syncthreads();
#pragma unroll
    for (int j = 0; j < 32; j += 8)
        WT[(size_t)(c0 + ty + j) * R + r0 + tx] = (bf16)tile[tx][ty + j];
}

// ---------------------------------------------------------------------------
// Row-major fragment load (used for both A and transposed-B operands).
// 16-bit 16x32 layout: lanes 0-15 -> row = lane, half-lane selects K octet:
//   elem e<8 : K = kb + e ; e>=8 : K = 16 + kb + (e-8), kb = (lane>>4)*8
// Two 128-bit loads per lane, concatenated in registers.
// ---------------------------------------------------------------------------
__device__ __forceinline__ v16bf load_frag(const bf16* __restrict__ P,
                                           int ld, int r0, int k0, int lane) {
    const int r  = r0 + (lane & 15);
    const int kb = (lane >> 4) << 3;
    const bf16* p = P + (size_t)r * ld + k0 + kb;
    v8bf lo = *(const v8bf*)p;          // K = kb + 0..7
    v8bf hi = *(const v8bf*)(p + 16);   // K = kb + 16..23
    return __builtin_shufflevector(lo, hi, 0, 1, 2, 3, 4, 5, 6, 7,
                                           8, 9, 10, 11, 12, 13, 14, 15);
}

// ---------------------------------------------------------------------------
// C = A(MxK,bf16,row) @ B(KxN) -> f32 row-major, with B given TRANSPOSED
// (BT is N x K row-major, so both operands load with contiguous-K b128s).
// One wave computes a 32(M) x 64(N) tile: 2 A-frags x 4 B-frags = 8 WMMAs
// per 32-wide K-step from 12 x 16B loads. Block = 128 (4 waves);
// grid = (N/256, M/32).
// ---------------------------------------------------------------------------
__global__ void __launch_bounds__(128)
gemm_bf16_wmma_kernel(const bf16* __restrict__ A, const bf16* __restrict__ BT,
                      float* __restrict__ C, int M, int N, int K) {
    const int lane = threadIdx.x & 31;
    const int wave = threadIdx.x >> 5;
    const int m0 = blockIdx.y * 32;
    const int n0 = (blockIdx.x * 4 + wave) * 64;
    if (m0 >= M || n0 >= N) return;   // wave-uniform: EXEC stays all-ones

    v8f acc[2][4];
#pragma unroll
    for (int i = 0; i < 2; ++i)
#pragma unroll
        for (int j = 0; j < 4; ++j) acc[i][j] = (v8f){};

    for (int k0 = 0; k0 < K; k0 += 32) {
        v16bf a0 = load_frag(A, K, m0,      k0, lane);
        v16bf a1 = load_frag(A, K, m0 + 16, k0, lane);
#pragma unroll
        for (int nt = 0; nt < 4; ++nt) {
            v16bf b = load_frag(BT, K, n0 + nt * 16, k0, lane);
            acc[0][nt] = __builtin_amdgcn_wmma_f32_16x16x32_bf16(
                false, a0, false, b, (short)0, acc[0][nt], false, false);
            acc[1][nt] = __builtin_amdgcn_wmma_f32_16x16x32_bf16(
                false, a1, false, b, (short)0, acc[1][nt], false, false);
        }
    }

    // C/D layout: VGPR r -> M = r + (lane>=16 ? 8 : 0), N = lane&15
    const int mrow = (lane >> 4) << 3;
    const int nc   = lane & 15;
#pragma unroll
    for (int mt = 0; mt < 2; ++mt) {
        float* cp = C + (size_t)(m0 + mt * 16 + mrow) * N + n0 + nc;
#pragma unroll
        for (int r = 0; r < 8; ++r) {
#pragma unroll
            for (int nt = 0; nt < 4; ++nt)
                cp[(size_t)r * N + nt * 16] = acc[mt][nt][r];
        }
    }
}

// ---------------------------------------------------------------------------
// gk = log_sigmoid((x @ Wgk1) @ Wgk2 + bgk2) / 16     (one block per token)
// ---------------------------------------------------------------------------
__global__ void __launch_bounds__(256)
gk_lowrank_kernel(const float* __restrict__ x, const float* __restrict__ W1,
                  const float* __restrict__ W2, const float* __restrict__ b2,
                  float* __restrict__ GK) {
    const int tok = blockIdx.x;
    const float* xr = x + (size_t)tok * HID;
    __shared__ float part[256];
    __shared__ float tsh[RANK];

    const int tid = threadIdx.x;
    const int r = tid >> 4, l16 = tid & 15;
    float s = 0.f;
    for (int i = l16; i < HID; i += 16) s += xr[i] * W1[(size_t)i * RANK + r];
    part[tid] = s;
    __syncthreads();
    if (tid < RANK) {
        float acc = 0.f;
#pragma unroll
        for (int j = 0; j < 16; ++j) acc += part[tid * 16 + j];
        tsh[tid] = acc;
    }
    __syncthreads();
#pragma unroll
    for (int j = 0; j < 4; ++j) {
        int o = tid + 256 * j;
        float s2 = b2[o];
#pragma unroll
        for (int rr = 0; rr < RANK; ++rr) s2 += tsh[rr] * W2[(size_t)rr * HID + o];
        // log_sigmoid(s) = min(s,0) - log1p(exp(-|s|))
        float ls = fminf(s2, 0.f) - log1pf(__expf(-fabsf(s2)));
        GK[(size_t)tok * HID + o] = ls * 0.0625f;
    }
}

// ---------------------------------------------------------------------------
// Gated linear attention scan. One block per (b,h); 256 threads.
// Q/K/V live in the fused QKVG buffer ([tok][4096], col 0/1024/2048).
// state[64][64] in registers: thread t owns row = t>>2, cols (t&3)*16..+15.
// ---------------------------------------------------------------------------
__global__ void __launch_bounds__(256)
gla_scan_kernel(const float* __restrict__ QKVG, const float* __restrict__ GK,
                float* __restrict__ O) {
    const int bh = blockIdx.x;
    const int b = bh / NH, h = bh % NH;
    __shared__ float qs[HD], ks[HD], vs[HD], gs[HD];
    __shared__ float red[HD * 4];

    const int t = threadIdx.x;
    const int row = t >> 2, cg = t & 3;
    float st[16];
#pragma unroll
    for (int j = 0; j < 16; ++j) st[j] = 0.f;

    const int which = t >> 6, d = t & 63;
    const size_t tok0 = (size_t)b * LL;

    for (int tt = 0; tt < LL; ++tt) {
        const size_t rb = (tok0 + tt) * N4;      // fused row base
        if      (which == 0) qs[d] = QKVG[rb + h * HD + d];
        else if (which == 1) ks[d] = QKVG[rb + HID + h * HD + d];
        else if (which == 2) vs[d] = QKVG[rb + 2 * HID + h * HD + d];
        else                 gs[d] = GK[(tok0 + tt) * HID + h * HD + d];
        __syncthreads();

        const float gate = __expf(gs[row]);
        const float vv = vs[row];
        float acc = 0.f;
#pragma unroll
        for (int j = 0; j < 16; ++j) {
            const int col = (cg << 4) + j;
            st[j] = st[j] * gate + vv * ks[col];
            acc += st[j] * qs[col];
        }
        red[row * 4 + cg] = acc;
        __syncthreads();
        if (t < HD) {
            float s = red[t * 4] + red[t * 4 + 1] + red[t * 4 + 2] + red[t * 4 + 3];
            O[(tok0 + tt) * HID + h * HD + t] = s;
        }
        __syncthreads();
    }
}

// ---------------------------------------------------------------------------
// Fused LayerNorm + SiLU output gate; emits bf16 for the final WMMA GEMM.
// Gate pre-activation lives at column 3072 of the fused QKVG buffer.
// One block per token.
// ---------------------------------------------------------------------------
__global__ void __launch_bounds__(256)
ln_gate_kernel(const float* __restrict__ AO, const float* __restrict__ QKVG,
               const float* __restrict__ gamma, const float* __restrict__ beta,
               bf16* __restrict__ Y) {
    const int tok = blockIdx.x;
    const float* o = AO + (size_t)tok * HID;
    const float* g = QKVG + (size_t)tok * N4 + 3 * HID;
    __shared__ float rs[256], rs2[256];

    const int tid = threadIdx.x;
    float s = 0.f, s2 = 0.f;
#pragma unroll
    for (int j = 0; j < 4; ++j) {
        float v = o[tid + 256 * j];
        s += v; s2 += v * v;
    }
    rs[tid] = s; rs2[tid] = s2;
    __syncthreads();
    for (int stp = 128; stp > 0; stp >>= 1) {
        if (tid < stp) { rs[tid] += rs[tid + stp]; rs2[tid] += rs2[tid + stp]; }
        __syncthreads();
    }
    const float mu = rs[0] * (1.f / HID);
    const float var = rs2[0] * (1.f / HID) - mu * mu;
    const float rstd = rsqrtf(var + 1e-5f);

#pragma unroll
    for (int j = 0; j < 4; ++j) {
        const int i = tid + 256 * j;
        float y = (o[i] - mu) * rstd * gamma[i] + beta[i];
        float gv = g[i];
        float silu = gv / (1.f + __expf(-gv));
        Y[(size_t)tok * HID + i] = (bf16)(y * silu);
    }
}

// ---------------------------------------------------------------------------
// launcher
// ---------------------------------------------------------------------------
extern "C" void kernel_launch(void* const* d_in, const int* in_sizes, int n_in,
                              void* d_out, int out_size, void* d_ws, size_t ws_size,
                              hipStream_t stream) {
    const float* x     = (const float*)d_in[0];
    const float* Wq    = (const float*)d_in[1];
    const float* Wk    = (const float*)d_in[2];
    const float* Wv    = (const float*)d_in[3];
    const float* Wg    = (const float*)d_in[4];
    const float* Wgk1  = (const float*)d_in[5];
    const float* Wgk2  = (const float*)d_in[6];
    const float* bgk2  = (const float*)d_in[7];
    const float* gamma = (const float*)d_in[8];
    const float* beta  = (const float*)d_in[9];
    const float* Wout  = (const float*)d_in[10];
    float* out = (float*)d_out;

    // workspace carve-up. WQT..WGT are CONSECUTIVE so together they form one
    // contiguous 4096 x 1024 bf16 B^T for the fused projection GEMM.
    char* p = (char*)d_ws;
    bf16* XB  = (bf16*)p;  p += (size_t)NTOK * HID * 2;       // x bf16
    bf16* WQT = (bf16*)p;  p += (size_t)HID * HID * 2;        // W^T bf16 (x4)
    bf16* WKT = (bf16*)p;  p += (size_t)HID * HID * 2;
    bf16* WVT = (bf16*)p;  p += (size_t)HID * HID * 2;
    bf16* WGT = (bf16*)p;  p += (size_t)HID * HID * 2;
    bf16* WOT = (bf16*)p;  p += (size_t)HID * HID * 2;
    float* QKVG = (float*)p; p += (size_t)NTOK * N4 * 4;      // fused Q|K|V|Gpre
    float* GK   = (float*)p; p += (size_t)NTOK * HID * 4;
    float* AO   = (float*)p; p += (size_t)NTOK * HID * 4;
    bf16* YB  = (bf16*)p;  p += (size_t)NTOK * HID * 2;

    const int nx = NTOK * HID;      // 4,194,304

    // 1) convert activations; transpose+convert weights (K x N -> N x K bf16)
    cvt_f32_bf16_kernel<<<(nx + 255) / 256, 256, 0, stream>>>(x, XB, nx);
    dim3 tgrid(HID / 32, HID / 32), tblk(32, 8);
    transpose_cvt_kernel<<<tgrid, tblk, 0, stream>>>(Wq,   WQT, HID, HID);
    transpose_cvt_kernel<<<tgrid, tblk, 0, stream>>>(Wk,   WKT, HID, HID);
    transpose_cvt_kernel<<<tgrid, tblk, 0, stream>>>(Wv,   WVT, HID, HID);
    transpose_cvt_kernel<<<tgrid, tblk, 0, stream>>>(Wg,   WGT, HID, HID);
    transpose_cvt_kernel<<<tgrid, tblk, 0, stream>>>(Wout, WOT, HID, HID);

    // 2) ONE fused projection GEMM: [4096x1024] @ [1024x4096] -> Q|K|V|Gpre.
    //    8192 waves in flight; single pass over XB.
    dim3 pgrid(N4 / 256, NTOK / 32);
    gemm_bf16_wmma_kernel<<<pgrid, 128, 0, stream>>>(XB, WQT, QKVG, NTOK, N4, HID);

    // 3) low-rank decay gates
    gk_lowrank_kernel<<<NTOK, 256, 0, stream>>>(x, Wgk1, Wgk2, bgk2, GK);

    // 4) sequential gated scan: one block per (b,h)
    gla_scan_kernel<<<BB * NH, 256, 0, stream>>>(QKVG, GK, AO);

    // 5) LayerNorm + SiLU gate -> bf16
    ln_gate_kernel<<<NTOK, 256, 0, stream>>>(AO, QKVG, gamma, beta, YB);

    // 6) output projection (WMMA)
    dim3 ogrid(HID / 256, NTOK / 32);
    gemm_bf16_wmma_kernel<<<ogrid, 128, 0, stream>>>(YB, WOT, out, NTOK, HID, HID);
}
